// EnhancedModel_66967130080000
// MI455X (gfx1250) — matrix operand order repaired and checked by reading, not compiled, not run on hardware
//
#include <hip/hip_runtime.h>
#include <hip/hip_bf16.h>

typedef __attribute__((ext_vector_type(16))) __bf16 v16bf;
typedef __attribute__((ext_vector_type(8)))  __bf16 v8bf;
typedef __attribute__((ext_vector_type(8)))  float  v8f;

constexpr int NB = 512;    // batch
constexpr int NT = 400;    // time
constexpr int NE = 32;     // embed
constexpr int NH = 256;    // hidden
constexpr int NL = 10;     // logits
constexpr int NG = 1024;   // 4*NH gate width

__device__ __forceinline__ float sigmoidf_(float x) {
  return 1.0f / (1.0f + __expf(-x));
}

// Assemble WMMA A operand (16x32 bf16) for this lane from a row-major row.
// Lane L (L<16): M=L, K = {0..15}; lane L+16: M=L, K = {16..31}.
__device__ __forceinline__ v16bf load_a16(const __bf16* rowBase, int kbA) {
  v8bf lo = *(const v8bf*)(rowBase + kbA);
  v8bf hi = *(const v8bf*)(rowBase + 8 + kbA);
  v16bf a;
#pragma unroll
  for (int i = 0; i < 8; ++i) { a[i] = lo[i]; a[i + 8] = hi[i]; }
  return a;
}

// ---------------- Phase 1a: build x = 0.5*mean(emb[tok0]) + 0.5*emb[tok2] (bf16)
__global__ void build_x_kernel(const int* __restrict__ tokens,
                               const float* __restrict__ emb,
                               __bf16* __restrict__ xbf) {
  const int gid  = blockIdx.x * blockDim.x + threadIdx.x;   // one lane per (b,t,e)
  const int bt   = gid >> 5;                                // one wave per (b,t)
  const int lane = gid & 31;                                // lane == e (E == 32)
  if (bt >= NB * NT) return;
  const int tok0 = tokens[bt * 3 + 0];
  const int tok2 = tokens[bt * 3 + 2];
  float s = emb[(size_t)tok0 * NE + lane];
#pragma unroll
  for (int m = 16; m >= 1; m >>= 1) s += __shfl_xor(s, m, 32);
  const float mean = s * (1.0f / 32.0f);
  const float xv = 0.5f * mean + 0.5f * emb[(size_t)tok2 * NE + lane];
  xbf[(size_t)bt * NE + lane] = (__bf16)xv;
}

// ---------------- Phase 1b: transpose+convert weights: dst[n][k] = (bf16)src[k][n]
__global__ void convT_kernel(const float* __restrict__ src, __bf16* __restrict__ dst,
                             int K, int N) {
  const int idx = blockIdx.x * blockDim.x + threadIdx.x;
  if (idx >= K * N) return;
  const int n = idx / K;
  const int k = idx - n * K;
  dst[(size_t)n * K + k] = (__bf16)src[(size_t)k * N + n];
}

// ---------------- Phase 2: persistent per-batch-tile bidirectional LSTM recurrence
// grid = (NB/16, 2 directions), block = 512 threads = 16 waves.
// Wave w owns hidden columns [16w, 16w+16): it computes the i/f/g/o gate tiles
// for those columns, keeps the cell state c in registers, and shares only h
// (bf16, 8 KB) through LDS between timesteps.
__global__ __launch_bounds__(512, 1) void lstm_rec_kernel(
    const __bf16* __restrict__ xbf,
    const __bf16* __restrict__ WTf, const __bf16* __restrict__ UTf, const float* __restrict__ bfv,
    const __bf16* __restrict__ WTb, const __bf16* __restrict__ UTb, const float* __restrict__ bbv,
    float* __restrict__ poolF, float* __restrict__ poolB) {
  const int dir = blockIdx.y;
  const __bf16* __restrict__ WT  = dir ? WTb : WTf;   // [NG][NE] bf16 (transposed)
  const __bf16* __restrict__ UT  = dir ? UTb : UTf;   // [NG][NH] bf16 (transposed)
  const float*  __restrict__ bia = dir ? bbv : bfv;   // [NG]
  float* __restrict__ pool       = dir ? poolB : poolF;

  __shared__ __align__(16) __bf16 lds_h[16 * NH];
  __shared__ __align__(16) __bf16 lds_x[16 * NE];
  __shared__ float lds_pool[16][16];

  const int tid  = threadIdx.x;
  const int wave = tid >> 5;        // 0..15, owns h-cols [16*wave, 16*wave+16)
  const int lane = tid & 31;
  const int l16  = lane & 15;
  const int half = lane >> 4;       // 0/1
  const int kbA  = half ? 16 : 0;   // A-operand K base per half-wave
  const int kbB  = half ? 16 : 0;   // B-operand K base per half-wave
  const int b0   = blockIdx.x * 16;

  for (int i = tid; i < 16 * NH; i += 512) lds_h[i] = (__bf16)0.0f;

  // Hoist W B-tiles and biases for this wave's 4 gate tiles (reused 400 steps;
  // only 32 VGPRs, worth keeping resident).
  v16bf wtile[4];
  float bgate[4];
#pragma unroll
  for (int G = 0; G < 4; ++G) {
    const int n = G * NH + wave * 16 + l16;
    wtile[G] = *(const v16bf*)(WT + (size_t)n * NE + kbB);
    bgate[G] = bia[n];
  }

  // Per-lane base of this wave's U columns: UT + n*NH + kbB, n = G*NH + 16w + l16.
  const __bf16* ubase[4];
#pragma unroll
  for (int G = 0; G < 4; ++G)
    ubase[G] = UT + (size_t)(G * NH + wave * 16 + l16) * NH + kbB;

  v8f c;
#pragma unroll
  for (int j = 0; j < 8; ++j) c[j] = 0.0f;

  __syncthreads();

  for (int t = 0; t < NT; ++t) {
    const int tt = dir ? (NT - 1 - t) : t;

    // Opaque zero byte-offset: makes U-tile addresses loop-variant so LICM
    // cannot hoist 256 VGPRs of loop-invariant loads (which spilled to scratch
    // last round). Keeps them as in-loop global_load_b128 served from L2/WGP$.
    int uoff = 0;
    asm volatile("" : "+v"(uoff));

    // Stage x tile (16 rows x 32 cols bf16): one element per thread.
    {
      const int m = tid >> 5;
      const int e = tid & 31;
      lds_x[m * NE + e] = xbf[((size_t)(b0 + m) * NT + tt) * NE + e];
      if (t + 1 < NT) {
        const int tn = dir ? (NT - 2 - t) : (t + 1);
        __builtin_prefetch(xbf + ((size_t)(b0 + m) * NT + tn) * NE + e, 0, 0);
      }
    }
    __syncthreads();

    // Accumulate z - bias; bias is added in the gate epilogue.
    v8f acc[4];
#pragma unroll
    for (int G = 0; G < 4; ++G)
#pragma unroll
      for (int j = 0; j < 8; ++j) acc[G][j] = 0.0f;

    // z += x_t @ W   (K = E = 32 -> exactly one bf16 WMMA per gate)
    {
      const v16bf ax = load_a16(lds_x + l16 * NE, kbA);
#pragma unroll
      for (int G = 0; G < 4; ++G)
        acc[G] = __builtin_amdgcn_wmma_f32_16x16x32_bf16(
            false, ax, false, wtile[G], (short)0, acc[G], false, false);
    }

    // z += h @ U     (K = H = 256 -> 8 chunks of 32, A shared across gates)
#pragma unroll
    for (int kc = 0; kc < 8; ++kc) {
      const v16bf ah = load_a16(lds_h + l16 * NH + kc * 32, kbA);
#pragma unroll
      for (int G = 0; G < 4; ++G) {
        const v16bf bu = *(const v16bf*)((const char*)(ubase[G] + kc * 32) + uoff);
        acc[G] = __builtin_amdgcn_wmma_f32_16x16x32_bf16(
            false, ah, false, bu, (short)0, acc[G], false, false);
      }
    }

    __syncthreads();  // all lds_h / lds_x reads for step t complete

    // Wave-local LSTM cell update (c never leaves registers).
    float hval[8];
#pragma unroll
    for (int j = 0; j < 8; ++j) {
      const float iv = sigmoidf_(acc[0][j] + bgate[0]);
      const float fv = sigmoidf_(acc[1][j] + bgate[1]);
      const float gv = tanhf(acc[2][j] + bgate[2]);
      const float ov = sigmoidf_(acc[3][j] + bgate[3]);
      const float cv = fv * c[j] + iv * gv;
      c[j] = cv;
      hval[j] = ov * tanhf(cv);
    }

    // Write h (bf16) for next step; lane-shuffle row-max for the pooling.
    const int ncol = wave * 16 + l16;
#pragma unroll
    for (int j = 0; j < 8; ++j) {
      const int m = j + 8 * half;                    // C/D layout: M per half-wave
      lds_h[m * NH + ncol] = (__bf16)hval[j];
      float r = hval[j];
#pragma unroll
      for (int mask = 1; mask <= 8; mask <<= 1)
        r = fmaxf(r, __shfl_xor(r, mask, 32));       // max over this wave's 16 cols
      if (l16 == 0) lds_pool[wave][m] = r;
    }
    __syncthreads();  // h + pool partials visible

    if (tid < 16) {
      float mx = lds_pool[0][tid];
#pragma unroll
      for (int w = 1; w < 16; ++w) mx = fmaxf(mx, lds_pool[w][tid]);
      pool[(size_t)(b0 + tid) * NT + tt] = mx;       // max over this direction's H
    }
  }
}

// ---------------- Phase 3: pooled = max(poolF, poolB); out = pooled @ Wd + bd
__global__ void proj_kernel(const float* __restrict__ poolF, const float* __restrict__ poolB,
                            const float* __restrict__ Wd, const float* __restrict__ bd,
                            float* __restrict__ out) {
  const int idx = blockIdx.x * blockDim.x + threadIdx.x;
  if (idx >= NB * NL) return;
  const int b = idx / NL;
  const int l = idx - b * NL;
  float acc = bd[l];
  for (int t = 0; t < NT; ++t) {
    const float p = fmaxf(poolF[(size_t)b * NT + t], poolB[(size_t)b * NT + t]);
    acc += p * Wd[t * NL + l];
  }
  out[idx] = acc;
}

extern "C" void kernel_launch(void* const* d_in, const int* in_sizes, int n_in,
                              void* d_out, int out_size, void* d_ws, size_t ws_size,
                              hipStream_t stream) {
  (void)in_sizes; (void)n_in; (void)out_size; (void)ws_size;
  const int*   tokens = (const int*)  d_in[0];
  const float* emb    = (const float*)d_in[1];
  const float* Wf     = (const float*)d_in[2];
  const float* Uf     = (const float*)d_in[3];
  const float* bfv    = (const float*)d_in[4];
  const float* Wb     = (const float*)d_in[5];
  const float* Ub     = (const float*)d_in[6];
  const float* bbv    = (const float*)d_in[7];
  const float* Wd     = (const float*)d_in[8];
  const float* bd     = (const float*)d_in[9];
  float* out = (float*)d_out;

  char* ws = (char*)d_ws;
  size_t off = 0;
  __bf16* xbf = (__bf16*)(ws + off); off += (size_t)NB * NT * NE * 2;  // 13.1 MB
  __bf16* WfT = (__bf16*)(ws + off); off += (size_t)NG * NE * 2;
  __bf16* WbT = (__bf16*)(ws + off); off += (size_t)NG * NE * 2;
  __bf16* UfT = (__bf16*)(ws + off); off += (size_t)NG * NH * 2;
  __bf16* UbT = (__bf16*)(ws + off); off += (size_t)NG * NH * 2;
  float* poolF = (float*)(ws + off); off += (size_t)NB * NT * 4;
  float* poolB = (float*)(ws + off); off += (size_t)NB * NT * 4;      // ~16 MB total

  build_x_kernel<<<(NB * NT * NE) / 256, 256, 0, stream>>>(tokens, emb, xbf);
  convT_kernel<<<(NE * NG + 255) / 256, 256, 0, stream>>>(Wf, WfT, NE, NG);
  convT_kernel<<<(NE * NG + 255) / 256, 256, 0, stream>>>(Wb, WbT, NE, NG);
  convT_kernel<<<(NH * NG + 255) / 256, 256, 0, stream>>>(Uf, UfT, NH, NG);
  convT_kernel<<<(NH * NG + 255) / 256, 256, 0, stream>>>(Ub, UbT, NH, NG);

  lstm_rec_kernel<<<dim3(NB / 16, 2), 512, 0, stream>>>(
      xbf, WfT, UfT, bfv, WbT, UbT, bbv, poolF, poolB);

  proj_kernel<<<(NB * NL + 255) / 256, 256, 0, stream>>>(poolF, poolB, Wd, bd, out);
}
